// RotarySelfAttention_79955111182470
// MI455X (gfx1250) — compile-verified, matrix operand order checked by reading, not run on hardware
//
#include <hip/hip_runtime.h>

// ---------------------------------------------------------------------------
// RotarySelfAttention for MI455X (gfx1250): bf16 WMMA pipeline.
// B=4 S=2048 D=1024 H=16 HD=64. Compute-bound -> v_wmma_f32_16x16x32_bf16.
// GEMM A-tiles staged via Tensor Data Mover (tensor_load_to_lds, TENSORcnt);
// attention K blocks staged via global_load_async_to_lds_b128 (ASYNCcnt),
// double-buffered against WMMA compute.
// ---------------------------------------------------------------------------

#define USE_ASYNC 1
#define USE_TDM 1

typedef __bf16    v16bf __attribute__((ext_vector_type(16)));
typedef float     v8f   __attribute__((ext_vector_type(8)));
typedef unsigned  u32x4 __attribute__((ext_vector_type(4)));
typedef unsigned  u32x8 __attribute__((ext_vector_type(8)));

#define NB 4
#define NS 2048
#define ND 1024
#define NH 16
#define NHD 64
#define NM (NB * NS)          // 8192 rows

static __device__ inline __bf16 f2bf(float f) {
  union { float f; unsigned u; } a; a.f = f;
  unsigned r = a.u + 0x7FFFu + ((a.u >> 16) & 1u);   // round-to-nearest-even
  unsigned short h = (unsigned short)(r >> 16);
  return __builtin_bit_cast(__bf16, h);
}

static __device__ inline v16bf ldv(const __bf16* p) {
  return *reinterpret_cast<const v16bf*>(p);
}

// Copy 16 bytes (8 bf16) global -> LDS (per-lane async, tracked by ASYNCcnt).
static __device__ inline void cp16(const __bf16* g, __bf16* l) {
#if USE_ASYNC
  unsigned lo = (unsigned)(unsigned long long)l;   // low 32 bits = LDS offset
  asm volatile("global_load_async_to_lds_b128 %0, %1, off"
               :: "v"(lo), "v"(g) : "memory");
#else
  *(uint4*)l = *(const uint4*)g;
#endif
}

static __device__ inline void cp_wait() {
#if USE_ASYNC
  asm volatile("s_wait_asynccnt 0" ::: "memory");
#endif
}

#if USE_TDM
// 1-D contiguous TDM copy: nelem bf16 elements, global -> LDS.
// D# per CDNA5 ISA ch8: group0 = {count/type/addrs}, group1 = dims/strides.
static __device__ inline void tdm_load_1d(unsigned lds_off, const void* gptr,
                                          unsigned nelem) {
  unsigned long long ga = (unsigned long long)gptr;
  u32x4 g0;
  g0.x = 1u;                                                   // count=1, user D#
  g0.y = __builtin_amdgcn_readfirstlane(lds_off);              // lds_addr
  g0.z = __builtin_amdgcn_readfirstlane((unsigned)ga);         // global_addr lo
  g0.w = (__builtin_amdgcn_readfirstlane((unsigned)(ga >> 32)) & 0x01FFFFFFu)
         | (2u << 30);                                         // addr hi | type=2
  u32x8 g1;
  g1[0] = 1u << 16;            // workgroup_mask=0, data_size=1 (2 bytes)
  g1[1] = nelem << 16;         // tensor_dim0[15:0] in bits[31:16]
  g1[2] = 1u << 16;            // tensor_dim0 hi = 0, tensor_dim1 = 1
  g1[3] = nelem << 16;         // tile_dim0
  g1[4] = 0u;                  // tile_dim1/tile_dim2 unused
  g1[5] = nelem;               // tensor_dim0_stride lo
  g1[6] = 0u;
  g1[7] = 0u;
  asm volatile("tensor_load_to_lds %0, %1" :: "s"(g0), "s"(g1) : "memory");
}
#endif

static __device__ inline v8f wmma_bf16(v16bf a, v16bf b, v8f c) {
  return __builtin_amdgcn_wmma_f32_16x16x32_bf16(false, a, false, b,
                                                 (short)0, c, false, false);
}

// ---------------------------------------------------------------------------
// Kernel 1: fp32 -> bf16
// ---------------------------------------------------------------------------
__global__ __launch_bounds__(256) void cvt_f32_bf16(const float* __restrict__ s,
                                                    __bf16* __restrict__ d, int n) {
  for (int i = blockIdx.x * blockDim.x + threadIdx.x; i < n;
       i += gridDim.x * blockDim.x)
    d[i] = f2bf(s[i]);
}

// ---------------------------------------------------------------------------
// Kernel 2: fused QKV projection + RoPE.
// grid = (512, 6), block = 256 (8 waves). Block: one 16-row x tile (TDM-staged
// into LDS), 512 output columns; each wave 16x64 outputs = 4 WMMA accumulators
// sharing one A-fragment per k-step (4x A-reuse).
// ---------------------------------------------------------------------------
__global__ __launch_bounds__(256) void qkv_rope_kernel(
    const __bf16* __restrict__ xb, const __bf16* __restrict__ w3,
    const float* __restrict__ bq, const float* __restrict__ bk,
    const float* __restrict__ bv,
    __bf16* __restrict__ qo, __bf16* __restrict__ ko, __bf16* __restrict__ vo) {
  __shared__ __bf16 sA[16 * ND];                       // 32 KB x tile

  const int tid  = threadIdx.x;
  const int lane = tid & 31, wv = tid >> 5;
  const int lrow = lane & 15, lhi = lane >> 4;
  const int mbase = blockIdx.x * 16;

  const __bf16* src = xb + (size_t)mbase * ND;
#if USE_TDM
  if (wv == 0) {
    tdm_load_1d((unsigned)(unsigned long long)&sA[0], src, 16 * ND);
    __builtin_amdgcn_s_wait_tensorcnt(0);
  }
#else
  #pragma unroll
  for (int j = 0; j < 8; ++j) {
    int c = (tid + 256 * j) * 8;
    cp16(src + c, sA + c);
  }
  cp_wait();
#endif
  __syncthreads();

  const int nb   = blockIdx.y * 512 + wv * 64;         // 0..3071, 64-aligned
  const int sect = nb >> 10;                           // 0:Q 1:K 2:V
  const int ncol = nb & 1023;
  const __bf16* W = w3 + (size_t)sect * ND * ND + (size_t)ncol * ND;

  v8f acc[4];
  #pragma unroll
  for (int t = 0; t < 4; ++t) acc[t] = (v8f){0, 0, 0, 0, 0, 0, 0, 0};

  for (int kk = 0; kk < ND; kk += 32) {
    v16bf a = ldv(sA + lrow * ND + kk + lhi * 16);     // shared A-frag
    #pragma unroll
    for (int t = 0; t < 4; ++t) {
      v16bf b = ldv(W + (size_t)(t * 16 + lrow) * ND + kk + lhi * 16);
      acc[t] = wmma_bf16(a, b, acc[t]);
    }
  }

  const int h  = ncol >> 6;                            // head (uniform per wave)
  const int bb = mbase >> 11;                          // batch (tile never spans)
  const int sb = mbase & (NS - 1);
  __bf16* dst = sect == 0 ? qo : (sect == 1 ? ko : vo);
  const float* bias = sect == 0 ? bq : (sect == 1 ? bk : bv);
  const bool doRope = sect < 2;

  #pragma unroll
  for (int t = 0; t < 4; ++t) {
    const int d = t * 16 + lrow;                       // dim within head
    const float bsv = bias[ncol + d];
    const float inv_freq = __expf(-0.14391156515f * (float)(d & ~1));
    #pragma unroll
    for (int e = 0; e < 8; ++e) {
      int row  = e + (lhi << 3);
      int spos = sb + row;
      float val = acc[t][e] + bsv;
      float other = __shfl_xor(val, 1, 32);            // RoPE pair partner
      float r = val;
      if (doRope) {
        float sn, cs;
        __sincosf((float)spos * inv_freq, &sn, &cs);
        r = (lane & 1) ? (other * sn + val * cs) : (val * cs - other * sn);
      }
      size_t idx = (((size_t)bb * NH + h) * NS + spos) * NHD + d;
      dst[idx] = f2bf(r);
    }
  }
}

// ---------------------------------------------------------------------------
// Kernel 3: causal flash attention, one (b, h, 64-query) tile per block.
// grid = B*H*(S/64) = 2048, block = 128 (4 waves x 16 query rows).
// K/V key-blocks double-buffered: async-stage block kb+1 while WMMAs run on kb.
// ---------------------------------------------------------------------------
__global__ __launch_bounds__(128) void attn_kernel(
    const __bf16* __restrict__ q, const __bf16* __restrict__ k,
    const __bf16* __restrict__ v, __bf16* __restrict__ ao) {
  __shared__ __bf16 sK[2][32 * 64];                    // key blocks, row-major
  __shared__ __bf16 sVt[2][64 * 32];                   // value blocks, transposed
  __shared__ __bf16 sP[4 * 16 * 32];                   // per-wave P tiles

  const int tid = threadIdx.x, lane = tid & 31, wv = tid >> 5;
  const int lrow = lane & 15, lhi = lane >> 4;
  const int qt = blockIdx.x & 31;
  const int hh = (blockIdx.x >> 5) & 15;
  const int bb = blockIdx.x >> 9;
  const size_t headoff = ((size_t)bb * NH + hh) * NS * NHD;
  const int qbase = qt * 64 + wv * 16;

  // Q fragments (16x64 per wave = two 16x32 A-frags), kept in registers.
  v16bf qf0 = ldv(q + headoff + (size_t)(qbase + lrow) * NHD + lhi * 16);
  v16bf qf1 = ldv(q + headoff + (size_t)(qbase + lrow) * NHD + 32 + lhi * 16);

  float m[8], l[8];
  v8f o[4];
  #pragma unroll
  for (int e = 0; e < 8; ++e) { m[e] = -1e30f; l[e] = 0.f; }
  #pragma unroll
  for (int t = 0; t < 4; ++t) o[t] = (v8f){0, 0, 0, 0, 0, 0, 0, 0};

  const int nkb = qt * 2 + 2;                          // key blocks of 32

  // Stage key-block kb into buffer `buf`: K flat (async), V transposed (VALU).
  auto stage = [&](int kb, int buf) {
    const int kstart = kb * 32;
    const __bf16* ksrc = k + headoff + (size_t)kstart * NHD;
    cp16(ksrc + tid * 8, &sK[buf][0] + tid * 8);
    cp16(ksrc + (tid + 128) * 8, &sK[buf][0] + (tid + 128) * 8);
    const __bf16* vsrc = v + headoff + (size_t)kstart * NHD;
    #pragma unroll
    for (int j = 0; j < 16; ++j) {
      int lin = tid * 16 + j;
      int key = lin >> 6, dd = lin & 63;
      sVt[buf][dd * 32 + key] = vsrc[lin];
    }
  };

  stage(0, 0);

  for (int kb = 0; kb < nkb; ++kb) {
    const int cur = kb & 1;
    const int kstart = kb * 32;
    cp_wait();                                         // K(cur) landed in LDS
    __syncthreads();                                   // + Vt(cur) visible
    if (kb + 1 < nkb) stage(kb + 1, cur ^ 1);          // prefetch next block

    if (kstart < qbase + 16) {                         // wave-uniform causal skip
      // S = Q K^T / sqrt(64): two 16x16 frags over 32 keys.
      v8f sf[2];
      #pragma unroll
      for (int half = 0; half < 2; ++half) {
        v16bf b0 = ldv(&sK[cur][0] + (half * 16 + lrow) * 64 + lhi * 16);
        v16bf b1 = ldv(&sK[cur][0] + (half * 16 + lrow) * 64 + 32 + lhi * 16);
        v8f sacc = {0.f, 0.f, 0.f, 0.f, 0.f, 0.f, 0.f, 0.f};
        sacc = wmma_bf16(qf0, b0, sacc);
        sacc = wmma_bf16(qf1, b1, sacc);
        sf[half] = sacc;
      }
      // Online softmax over key dimension (16 lanes of a half-wave = one row).
      #pragma unroll
      for (int e = 0; e < 8; ++e) {
        int row = e + (lhi << 3);
        int qi = qbase + row;
        float x0 = sf[0][e] * 0.125f;
        float x1 = sf[1][e] * 0.125f;
        if (kstart + lrow > qi)      x0 = -1e30f;
        if (kstart + 16 + lrow > qi) x1 = -1e30f;
        float rm = fmaxf(x0, x1);
        rm = fmaxf(rm, __shfl_xor(rm, 1, 32));
        rm = fmaxf(rm, __shfl_xor(rm, 2, 32));
        rm = fmaxf(rm, __shfl_xor(rm, 4, 32));
        rm = fmaxf(rm, __shfl_xor(rm, 8, 32));
        float mn = fmaxf(m[e], rm);
        float al = __expf(m[e] - mn);
        float p0 = __expf(x0 - mn);
        float p1 = __expf(x1 - mn);
        float rs = p0 + p1;
        rs += __shfl_xor(rs, 1, 32);
        rs += __shfl_xor(rs, 2, 32);
        rs += __shfl_xor(rs, 4, 32);
        rs += __shfl_xor(rs, 8, 32);
        l[e] = l[e] * al + rs;
        m[e] = mn;
        #pragma unroll
        for (int t = 0; t < 4; ++t) o[t][e] *= al;
        sP[wv * 512 + row * 32 + lrow]      = f2bf(p0);
        sP[wv * 512 + row * 32 + 16 + lrow] = f2bf(p1);
      }
      // Wave-private LDS round-trip: wait only on this wave's DS ops.
      asm volatile("s_wait_dscnt 0" ::: "memory");
      v16bf pf = ldv(sP + wv * 512 + lrow * 32 + lhi * 16);
      #pragma unroll
      for (int t = 0; t < 4; ++t) {
        v16bf bV = ldv(&sVt[cur][0] + (t * 16 + lrow) * 32 + lhi * 16);
        o[t] = wmma_bf16(pf, bV, o[t]);
      }
    }
  }

  // Epilogue: normalize and store to (B, S, H*HD) bf16.
  #pragma unroll
  for (int e = 0; e < 8; ++e) {
    int row = e + (lhi << 3);
    float invl = 1.f / l[e];
    int spos = qbase + row;
    #pragma unroll
    for (int t = 0; t < 4; ++t) {
      int d = t * 16 + lrow;
      size_t idx = ((size_t)bb * NS + spos) * ND + hh * NHD + d;
      ao[idx] = f2bf(o[t][e] * invl);
    }
  }
}

// ---------------------------------------------------------------------------
// Kernel 4: output projection, fp32 result + bias.
// grid = (512, 2), block = 256 (8 waves), 16x64 outputs per wave.
// ---------------------------------------------------------------------------
__global__ __launch_bounds__(256) void oproj_kernel(
    const __bf16* __restrict__ ao, const __bf16* __restrict__ wo,
    const float* __restrict__ bo, float* __restrict__ out) {
  __shared__ __bf16 sA[16 * ND];

  const int tid = threadIdx.x;
  const int lane = tid & 31, wv = tid >> 5;
  const int lrow = lane & 15, lhi = lane >> 4;
  const int mbase = blockIdx.x * 16;

  const __bf16* src = ao + (size_t)mbase * ND;
#if USE_TDM
  if (wv == 0) {
    tdm_load_1d((unsigned)(unsigned long long)&sA[0], src, 16 * ND);
    __builtin_amdgcn_s_wait_tensorcnt(0);
  }
#else
  #pragma unroll
  for (int j = 0; j < 8; ++j) {
    int c = (tid + 256 * j) * 8;
    cp16(src + c, sA + c);
  }
  cp_wait();
#endif
  __syncthreads();

  const int nb = blockIdx.y * 512 + wv * 64;

  v8f acc[4];
  #pragma unroll
  for (int t = 0; t < 4; ++t) acc[t] = (v8f){0, 0, 0, 0, 0, 0, 0, 0};

  for (int kk = 0; kk < ND; kk += 32) {
    v16bf a = ldv(sA + lrow * ND + kk + lhi * 16);
    #pragma unroll
    for (int t = 0; t < 4; ++t) {
      v16bf b = ldv(wo + (size_t)(nb + t * 16 + lrow) * ND + kk + lhi * 16);
      acc[t] = wmma_bf16(a, b, acc[t]);
    }
  }

  #pragma unroll
  for (int t = 0; t < 4; ++t) {
    const int col = nb + t * 16 + lrow;
    const float bias = bo[col];
    #pragma unroll
    for (int e = 0; e < 8; ++e) {
      int row = e + (lhi << 3);
      out[(size_t)(mbase + row) * ND + col] = acc[t][e] + bias;
    }
  }
}

// ---------------------------------------------------------------------------
// Host launcher.
// ---------------------------------------------------------------------------
extern "C" void kernel_launch(void* const* d_in, const int* in_sizes, int n_in,
                              void* d_out, int out_size, void* d_ws, size_t ws_size,
                              hipStream_t stream) {
  const float* x  = (const float*)d_in[0];
  const float* Wq = (const float*)d_in[1];
  const float* bq = (const float*)d_in[2];
  const float* Wk = (const float*)d_in[3];
  const float* bk = (const float*)d_in[4];
  const float* Wv = (const float*)d_in[5];
  const float* bv = (const float*)d_in[6];
  const float* Wo = (const float*)d_in[7];
  const float* bo = (const float*)d_in[8];
  float* out = (float*)d_out;

  const size_t nX = (size_t)NM * ND;      // 8,388,608
  const size_t nW = (size_t)ND * ND;      // 1,048,576

  char* ws = (char*)d_ws;
  __bf16* xb  = (__bf16*)(ws);                               // 16 MiB
  __bf16* w3  = (__bf16*)(ws + nX * 2);                      // 6 MiB (Wq|Wk|Wv)
  __bf16* wob = (__bf16*)(ws + nX * 2 + 3 * nW * 2);         // 2 MiB
  __bf16* qb  = (__bf16*)(ws + nX * 2 + 4 * nW * 2);         // 16 MiB
  __bf16* kb  = qb + nX;                                     // 16 MiB
  __bf16* vb  = kb + nX;                                     // 16 MiB
  __bf16* aob = vb + nX;                                     // 16 MiB

  cvt_f32_bf16<<<4096, 256, 0, stream>>>(x,  xb,          (int)nX);
  cvt_f32_bf16<<<1024, 256, 0, stream>>>(Wq, w3,          (int)nW);
  cvt_f32_bf16<<<1024, 256, 0, stream>>>(Wk, w3 + nW,     (int)nW);
  cvt_f32_bf16<<<1024, 256, 0, stream>>>(Wv, w3 + 2 * nW, (int)nW);
  cvt_f32_bf16<<<1024, 256, 0, stream>>>(Wo, wob,         (int)nW);

  qkv_rope_kernel<<<dim3(NM / 16, 3 * ND / 512), 256, 0, stream>>>(
      xb, w3, bq, bk, bv, qb, kb, vb);

  attn_kernel<<<NB * NH * (NS / 64), 128, 0, stream>>>(qb, kb, vb, aob);

  oproj_kernel<<<dim3(NM / 16, ND / 512), 256, 0, stream>>>(aob, wob, bo, out);
}
